// MultiHeadAttention_67645734912261
// MI455X (gfx1250) — matrix operand order repaired and checked by reading, not compiled, not run on hardware
//
#include <hip/hip_runtime.h>

// Problem constants (match reference)
#define B_  4
#define S_  2048
#define D_  1024
#define H_  16
#define HD_ 64

typedef __attribute__((ext_vector_type(16))) _Float16 v16h;
typedef __attribute__((ext_vector_type(8)))  _Float16 v8h;
typedef __attribute__((ext_vector_type(8)))  float    v8f;

__device__ __forceinline__ v8f wmma16(v16h a, v16h b, v8f c) {
  // v_wmma_f32_16x16x32_f16 : D = A(16x32 f16) x B(32x16 f16) + C(16x16 f32)
  return __builtin_amdgcn_wmma_f32_16x16x32_f16(false, a, false, b, (short)0, c,
                                                false, false);
}

__device__ __forceinline__ v8f zero8() { v8f z = {}; return z; }

// ---- A fragment: 16x32 f16, row-major source, leading dim `ld` (elements).
// ISA 16-bit A layout: lanes 0-15 row M=lane, K chunks {0..7,16..23};
// lanes 16-31 row M=lane-16, K chunks {8..15,24..31}.
__device__ __forceinline__ v16h load_fragA(const _Float16* base, int ld, int lane) {
  const _Float16* p = base + (size_t)(lane & 15) * ld + ((lane >> 4) * 8);
  v8h lo = *(const v8h*)(p);
  v8h hi = *(const v8h*)(p + 16);
  v16h r;
#pragma unroll
  for (int i = 0; i < 8; ++i) { r[i] = lo[i]; r[i + 8] = hi[i]; }
  return r;
}

// ---- B fragment: 32x16 f16. Source is Bt[N][K] row-major (ld = K).
// ISA 16-bit B layout: lanes 0-15 col N=lane, K chunks {0..7,16..23}; lanes 16-31 K chunks {8..15,24..31}.
__device__ __forceinline__ v16h load_fragB(const _Float16* base, int ld, int lane) {
  const _Float16* p = base + (size_t)(lane & 15) * ld + ((lane >> 4) * 8);
  v8h lo = *(const v8h*)(p);
  v8h hi = *(const v8h*)(p + 16);
  v16h r;
#pragma unroll
  for (int i = 0; i < 8; ++i) { r[i] = lo[i]; r[i + 8] = hi[i]; }
  return r;
}

// ---- 64x64 wave tile GEMM core: C = A[M][K] x Bt[N][K]^T, both f16, ld = K.
__device__ __forceinline__ void gemm_tile_64x64(const _Float16* A, const _Float16* Bt,
                                                int K, int m0, int n0, int lane,
                                                v8f acc[4][4]) {
#pragma unroll
  for (int mf = 0; mf < 4; ++mf)
#pragma unroll
    for (int nf = 0; nf < 4; ++nf) acc[mf][nf] = zero8();
  for (int k0 = 0; k0 < K; k0 += 32) {
    v16h a[4], b[4];
#pragma unroll
    for (int mf = 0; mf < 4; ++mf)
      a[mf] = load_fragA(A + (size_t)(m0 + mf * 16) * K + k0, K, lane);
#pragma unroll
    for (int nf = 0; nf < 4; ++nf)
      b[nf] = load_fragB(Bt + (size_t)(n0 + nf * 16) * K + k0, K, lane);
#pragma unroll
    for (int mf = 0; mf < 4; ++mf)
#pragma unroll
      for (int nf = 0; nf < 4; ++nf)
        acc[mf][nf] = wmma16(a[mf], b[nf], acc[mf][nf]);
  }
}

// ================= prep kernels =================
__global__ void cvt_f32_f16_kernel(const float* __restrict__ in,
                                   _Float16* __restrict__ out, int n) {
  int i = blockIdx.x * blockDim.x + threadIdx.x;
  if (i < n) out[i] = (_Float16)in[i];
}

// in[R][C] f32 -> out[C][R] f16
__global__ void transpose_cvt_kernel(const float* __restrict__ in,
                                     _Float16* __restrict__ out, int R, int C) {
  int idx = blockIdx.x * blockDim.x + threadIdx.x;
  if (idx >= R * C) return;
  int r = idx / C, c = idx - r * C;
  out[(size_t)c * R + r] = (_Float16)in[idx];
}

// ================= QKV projection =================
// C[8192][3072] = xh @ wqkvT^T + b_qkv, scattered into Q/K/Vt f16 buffers.
__global__ void __launch_bounds__(256)
qkv_gemm_kernel(const _Float16* __restrict__ xh, const _Float16* __restrict__ wqkvT,
                const float* __restrict__ b_qkv,
                _Float16* __restrict__ Q, _Float16* __restrict__ Km,
                _Float16* __restrict__ Vt) {
  int tid = threadIdx.x, lane = tid & 31, wv = tid >> 5;
  int m0 = blockIdx.y * 128 + (wv >> 2) * 64;
  int n0 = blockIdx.x * 256 + (wv & 3) * 64;
  v8f acc[4][4];
  gemm_tile_64x64(xh, wqkvT, D_, m0, n0, lane, acc);
  int hi8 = (lane >> 4) << 3, cl = lane & 15;
#pragma unroll
  for (int mf = 0; mf < 4; ++mf)
#pragma unroll
    for (int nf = 0; nf < 4; ++nf)
#pragma unroll
      for (int i = 0; i < 8; ++i) {
        int m = m0 + mf * 16 + i + hi8;          // global row = b*S + s
        int c = n0 + nf * 16 + cl;               // global col in [0,3D)
        float v = acc[mf][nf][i] + b_qkv[c];
        int t = c >> 10, rem = c & 1023, h = rem >> 6, hd = rem & 63;
        int b = m >> 11, s = m & 2047;
        _Float16 hv = (_Float16)v;
        size_t bh = (size_t)(b * H_ + h);
        if (t == 0)      Q [(bh * S_ + s) * HD_ + hd] = hv;
        else if (t == 1) Km[(bh * S_ + s) * HD_ + hd] = hv;
        else             Vt[(bh * HD_ + hd) * S_ + s] = hv;
      }
}

// ================= attention =================
// One block per (b, 16-row q tile); loops all heads -> deterministic head-mean.
// Dyn LDS: sc (16x2048 f16, 64KB) | redC (8x16x64 f32, 32KB) | red (256 f32, 1KB)
__global__ void __launch_bounds__(256)
attn_kernel(const _Float16* __restrict__ Q, const _Float16* __restrict__ Km,
            const _Float16* __restrict__ Vt, _Float16* __restrict__ attO,
            float* __restrict__ out_mean) {
  extern __shared__ char smem[];
  _Float16* sc  = (_Float16*)smem;                                   // 65536 B
  float*    redC = (float*)(smem + 16 * S_ * 2);                     // 32768 B
  float*    red  = (float*)(smem + 16 * S_ * 2 + 8 * 1024 * 4);      //  1024 B

  int tid = threadIdx.x, lane = tid & 31, wv = tid >> 5;
  int q0 = blockIdx.x * 16;
  int b  = blockIdx.y;
  int cl = lane & 15, hi8 = (lane >> 4) << 3;

  for (int h = 0; h < H_; ++h) {
    size_t bh = (size_t)(b * H_ + h);

    // ---- phase 1: scores = (Q Kt) / sqrt(HD) -> sc as f16 ----
    const _Float16* Qb = Q + (bh * S_ + q0) * HD_;
    v16h aq0 = load_fragA(Qb + 0,  HD_, lane);
    v16h aq1 = load_fragA(Qb + 32, HD_, lane);
    for (int nt = 0; nt < 16; ++nt) {
      int n0 = wv * 256 + nt * 16;
      const _Float16* Kb = Km + (bh * S_ + n0) * HD_;
      v8f acc = zero8();
      acc = wmma16(aq0, load_fragB(Kb + 0,  HD_, lane), acc);
      acc = wmma16(aq1, load_fragB(Kb + 32, HD_, lane), acc);
#pragma unroll
      for (int i = 0; i < 8; ++i)
        sc[(i + hi8) * S_ + n0 + cl] = (_Float16)(acc[i] * 0.125f);
    }
    __syncthreads();

    // ---- phase 2: softmax per row + head-mean accumulation ----
    {
      int r = tid >> 4, seg = tid & 15;     // tid == r*16 + seg
      _Float16* srow = sc + r * S_;
      int c0 = seg * 128;
      float lmax = -3.0e38f;
      for (int c = c0; c < c0 + 128; ++c) {
        float t = (float)srow[c];
        lmax = t > lmax ? t : lmax;
      }
      red[tid] = lmax;
      __syncthreads();
      float rmax = -3.0e38f;
#pragma unroll
      for (int j = 0; j < 16; ++j) {
        float t = red[(r << 4) + j];
        rmax = t > rmax ? t : rmax;
      }
      __syncthreads();
      float lsum = 0.f;
      for (int c = c0; c < c0 + 128; ++c) {
        float p = __expf((float)srow[c] - rmax);
        srow[c] = (_Float16)p;
        lsum += p;
      }
      red[tid] = lsum;
      __syncthreads();
      float rsum = 0.f;
#pragma unroll
      for (int j = 0; j < 16; ++j) rsum += red[(r << 4) + j];
      float inv = 1.0f / rsum;
      float* mrow = out_mean + ((size_t)b * S_ + q0 + r) * S_;
      for (int c = c0; c < c0 + 128; ++c) {
        float p = (float)srow[c] * inv;
        srow[c] = (_Float16)p;                 // normalized prob for P@V
        float mv = p * (1.0f / (float)H_);
        if (h == 0) mrow[c] = mv;              // first head: init (d_out poisoned)
        else        mrow[c] += mv;             // same thread owns (r,c): no race
      }
    }
    __syncthreads();

    // ---- phase 3: out_tile = P @ V (K split across 8 waves, LDS reduce) ----
    v8f oc[4];
#pragma unroll
    for (int nf = 0; nf < 4; ++nf) oc[nf] = zero8();
    int kb = wv * 256;
    for (int kk = kb; kk < kb + 256; kk += 32) {
      v16h a = load_fragA(sc + kk, S_, lane);              // from LDS
#pragma unroll
      for (int nf = 0; nf < 4; ++nf) {
        const _Float16* Vb = Vt + (bh * HD_ + nf * 16) * (size_t)S_ + kk;
        oc[nf] = wmma16(a, load_fragB(Vb, S_, lane), oc[nf]);
      }
    }
#pragma unroll
    for (int nf = 0; nf < 4; ++nf)
#pragma unroll
      for (int i = 0; i < 8; ++i)
        redC[wv * 1024 + (i + hi8) * 64 + nf * 16 + cl] = oc[nf][i];
    __syncthreads();
#pragma unroll
    for (int j = 0; j < 4; ++j) {
      int e = tid * 4 + j;                    // 1024 elements of 16x64 tile
      int row = e >> 6, col = e & 63;
      float s = 0.f;
#pragma unroll
      for (int w = 0; w < 8; ++w) s += redC[w * 1024 + e];
      attO[((size_t)b * S_ + q0 + row) * D_ + h * HD_ + col] = (_Float16)s;
    }
    __syncthreads();   // before next head overwrites sc/redC
  }
}

// ================= output projection =================
__global__ void __launch_bounds__(256)
proj_gemm_kernel(const _Float16* __restrict__ aO, const _Float16* __restrict__ wprojT,
                 const float* __restrict__ b_proj, float* __restrict__ out) {
  int tid = threadIdx.x, lane = tid & 31, wv = tid >> 5;
  int m0 = blockIdx.y * 128 + (wv >> 2) * 64;
  int n0 = blockIdx.x * 256 + (wv & 3) * 64;
  v8f acc[4][4];
  gemm_tile_64x64(aO, wprojT, D_, m0, n0, lane, acc);
  int hi8 = (lane >> 4) << 3, cl = lane & 15;
#pragma unroll
  for (int mf = 0; mf < 4; ++mf)
#pragma unroll
    for (int nf = 0; nf < 4; ++nf)
#pragma unroll
      for (int i = 0; i < 8; ++i) {
        int m = m0 + mf * 16 + i + hi8;
        int c = n0 + nf * 16 + cl;
        out[(size_t)m * D_ + c] = acc[mf][nf][i] + b_proj[c];
      }
}

// ================= launcher =================
extern "C" void kernel_launch(void* const* d_in, const int* in_sizes, int n_in,
                              void* d_out, int out_size, void* d_ws, size_t ws_size,
                              hipStream_t stream) {
  (void)in_sizes; (void)n_in; (void)out_size; (void)ws_size;
  const float* x      = (const float*)d_in[0];
  const float* w_qkv  = (const float*)d_in[1];
  const float* b_qkv  = (const float*)d_in[2];
  const float* w_proj = (const float*)d_in[3];
  const float* b_proj = (const float*)d_in[4];

  float* out      = (float*)d_out;                       // [B,S,D]
  float* out_mean = out + (size_t)B_ * S_ * D_;          // [B,S,S]

  // workspace layout (bytes); total 92,274,688 (~88 MB)
  char* ws = (char*)d_ws;
  _Float16* xh     = (_Float16*)(ws + 0);                 // 16 MB  [B*S][D]
  _Float16* wqkvT  = (_Float16*)(ws + (16u << 20));       //  6 MB  [3D][D]
  _Float16* wprojT = (_Float16*)(ws + 23068672u);         //  2 MB  [D][D]
  _Float16* Q      = (_Float16*)(ws + 25165824u);         // 16 MB  [B,H,S,HD]
  _Float16* Km     = (_Float16*)(ws + 41943040u);         // 16 MB  [B,H,S,HD]
  _Float16* Vt     = (_Float16*)(ws + 58720256u);         // 16 MB  [B,H,HD,S]
  _Float16* attO   = (_Float16*)(ws + 75497472u);         // 16 MB  [B*S][D]

  // 1) convert x to f16
  {
    int n = B_ * S_ * D_;
    cvt_f32_f16_kernel<<<n / 256, 256, 0, stream>>>(x, xh, n);
  }
  // 2) transpose+convert weights so B-matrix columns are contiguous
  {
    int n = D_ * 3 * D_;
    transpose_cvt_kernel<<<n / 256, 256, 0, stream>>>(w_qkv, wqkvT, D_, 3 * D_);
  }
  {
    int n = D_ * D_;
    transpose_cvt_kernel<<<n / 256, 256, 0, stream>>>(w_proj, wprojT, D_, D_);
  }
  // 3) fused QKV projection -> Q, K, Vt (f16)
  qkv_gemm_kernel<<<dim3(3 * D_ / 256, (B_ * S_) / 128), 256, 0, stream>>>(
      xh, wqkvT, b_qkv, Q, Km, Vt);
  // 4) attention (scores + softmax + head-mean + P@V)
  {
    size_t smem = (size_t)16 * S_ * 2 + 8 * 1024 * 4 + 256 * 4;  // 99,328 B
    attn_kernel<<<dim3(S_ / 16, B_), 256, smem, stream>>>(Q, Km, Vt, attO, out_mean);
  }
  // 5) output projection
  proj_gemm_kernel<<<dim3(D_ / 256, (B_ * S_) / 128), 256, 0, stream>>>(
      attO, wprojT, b_proj, out);
}
